// Transformer_57775900066384
// MI455X (gfx1250) — matrix operand order repaired and checked by reading, not compiled, not run on hardware
//
#include <hip/hip_runtime.h>

// ---------------- problem constants ----------------
#define BB   4
#define SS   512
#define DD   1024
#define HH   16
#define HDIM 64
#define EE   8
#define FFD  2048
#define LL   2
#define VT   32000
#define BSTOK (BB * SS)   // 2048 tokens

typedef __attribute__((ext_vector_type(16))) __bf16 v16bf;
typedef __attribute__((ext_vector_type(8)))  __bf16 v8bf;
typedef __attribute__((ext_vector_type(4)))  __bf16 v4bf;
typedef __attribute__((ext_vector_type(8)))  float  v8f;

// fp32 -> bf16 round-to-nearest-even
static __device__ __forceinline__ __bf16 f2bf(float f) {
    unsigned u = __builtin_bit_cast(unsigned, f);
    u += 0x7FFFu + ((u >> 16) & 1u);
    unsigned short h = (unsigned short)(u >> 16);
    return __builtin_bit_cast(__bf16, h);
}

// ---------------- one-shot fp32 -> bf16 conversion (vectorized) ----------------
__global__ __launch_bounds__(256)
void cvt_bf16_kernel(const float* __restrict__ in, __bf16* __restrict__ out, long n4)
{
    long i = (long)blockIdx.x * 256 + threadIdx.x;
    if (i >= n4) return;
    float4 f = reinterpret_cast<const float4*>(in)[i];
    v4bf r;
    r[0] = f2bf(f.x); r[1] = f2bf(f.y); r[2] = f2bf(f.z); r[3] = f2bf(f.w);
    reinterpret_cast<v4bf*>(out)[i] = r;
}

// ---------------------------------------------------------------------------
// Double-buffered LDS-staged batched GEMM via v_wmma_f32_16x16x32_bf16.
//   C[m,n] (+)= rowScale[m] * (relu?)(sum_k A[m,k]*B[k,n] + bias[n])
// Block tile: 128(M) x 64(N) x 32(K). 8 waves arranged 4(M) x 2(N); each wave
// owns a 32x32 quadrant = 2x2 WMMA tiles = 4 v_wmma per K-step.
// Pipeline: global loads for step k+1 issue before the WMMAs of step k;
// LDS ping-pong gives one barrier per K-step with no load->store->barrier
// chain in front of the matrix ops.
//   A frag: K = (e<8 ? e : e+8) + 8*half  -> two 8-elem runs -> 2x ds_load_b128
//   B frag: K = (e<8 ? e : e+8) + 8*half  -> two 8-elem runs -> 2x ds_load_b128
//   D frag: reg r -> M = r + 8*half, N = lane&15
// Requires M%128==0, N%64==0, K%32==0 (true for every GEMM in this model).
// ---------------------------------------------------------------------------
template <int BT>
__global__ __launch_bounds__(256)
void gemm_wmma_kernel(const __bf16* __restrict__ A, const __bf16* __restrict__ Bm,
                      const float* __restrict__ bias,
                      const float* __restrict__ rowScale, int rsStride,
                      float* __restrict__ C,
                      int M, int N, int K,
                      int lda, int ldb, int ldc,
                      int doRelu, int accum,
                      int batchH,
                      long sA0, long sA1, long sB0, long sB1, long sC0, long sC1)
{
    __shared__ __bf16 lds_a[2][128 * 32];   // [m][k]   2 x 8 KB
    __shared__ __bf16 lds_bt[2][64 * 32];   // [n][k]   2 x 4 KB

    const int tid  = threadIdx.x;
    const int lane = tid & 31;
    const int wv   = tid >> 5;
    const int tilesN = N >> 6;                 // N / 64
    const int tm = (int)blockIdx.x / tilesN;
    const int tn = (int)blockIdx.x % tilesN;
    const int m0 = tm * 128;
    const int c0 = tn * 64;

    const int z  = blockIdx.y;
    const int zb = z / batchH, zh = z % batchH;
    A  += zb * sA0 + zh * sA1;
    Bm += zb * sB0 + zh * sB1;
    C  += zb * sC0 + zh * sC1;

    const int l15  = lane & 15;
    const int half = lane >> 4;
    const int waveM = wv >> 1;                 // 0..3
    const int waveN = wv & 1;                  // 0..1

    // cooperative-load coordinates
    const int ar    = tid >> 1;                // A row 0..127
    const int ahalf = tid & 1;                 // which 16-elem half of the 32-wide K strip
    const int bn    = tid >> 2;                // (BT) B row (=output col) 0..63
    const int bq    = tid & 3;                 // (BT) 8-elem chunk
    const int bk    = tid >> 3;                // (!BT) B K-row 0..31
    const int bo    = tid & 7;                 // (!BT) 8-col chunk

    // staged global loads (register)
    v8bf av0, av1, bv;
    auto loadTile = [&](int k0) {
        const __bf16* Ap = A + (long)(m0 + ar) * lda + k0 + ahalf * 16;
        __builtin_prefetch(Ap + 64, 0, 1);
        av0 = *reinterpret_cast<const v8bf*>(Ap);
        av1 = *reinterpret_cast<const v8bf*>(Ap + 8);
        if (BT) {
            const __bf16* Bp = Bm + (long)(c0 + bn) * ldb + k0 + bq * 8;
            __builtin_prefetch(Bp + 64, 0, 1);
            bv = *reinterpret_cast<const v8bf*>(Bp);
        } else {
            const __bf16* Bp = Bm + (long)(k0 + bk) * ldb + c0 + bo * 8;
            __builtin_prefetch(Bp + (long)64 * ldb, 0, 1);
            bv = *reinterpret_cast<const v8bf*>(Bp);
        }
    };
    auto storeTile = [&](int buf) {
        *reinterpret_cast<v8bf*>(&lds_a[buf][ar * 32 + ahalf * 16])     = av0;
        *reinterpret_cast<v8bf*>(&lds_a[buf][ar * 32 + ahalf * 16 + 8]) = av1;
        if (BT) {
            *reinterpret_cast<v8bf*>(&lds_bt[buf][bn * 32 + bq * 8]) = bv;
        } else {
#pragma unroll
            for (int e = 0; e < 8; ++e)
                lds_bt[buf][(bo * 8 + e) * 32 + bk] = bv[e];
        }
    };

    v8f acc00 = {0.f,0.f,0.f,0.f,0.f,0.f,0.f,0.f};
    v8f acc01 = acc00, acc10 = acc00, acc11 = acc00;

    const int nsteps = K >> 5;                 // K / 32
    loadTile(0);
    storeTile(0);
    __syncthreads();

    for (int i = 0; i < nsteps; ++i) {
        const int cur = i & 1;
        const bool hasNext = (i + 1) < nsteps;
        if (hasNext) loadTile((i + 1) * 32);   // global loads overlap the WMMAs below

        // ---- fragments from LDS buf[cur] (128-bit ds loads) ----
        v16bf a0, a1, b0, b1;
        {
            const __bf16* p0 = &lds_a[cur][(waveM * 32 + l15) * 32 + half * 8];
            v8bf lo = *reinterpret_cast<const v8bf*>(p0);
            v8bf hi = *reinterpret_cast<const v8bf*>(p0 + 16);
            a0 = __builtin_shufflevector(lo, hi, 0,1,2,3,4,5,6,7,8,9,10,11,12,13,14,15);
            const __bf16* p1 = p0 + 16 * 32;
            lo = *reinterpret_cast<const v8bf*>(p1);
            hi = *reinterpret_cast<const v8bf*>(p1 + 16);
            a1 = __builtin_shufflevector(lo, hi, 0,1,2,3,4,5,6,7,8,9,10,11,12,13,14,15);
        }
        {
            const __bf16* p0 = &lds_bt[cur][(waveN * 32 + l15) * 32 + half * 8];
            v8bf lo = *reinterpret_cast<const v8bf*>(p0);
            v8bf hi = *reinterpret_cast<const v8bf*>(p0 + 16);
            b0 = __builtin_shufflevector(lo, hi, 0,1,2,3,4,5,6,7,8,9,10,11,12,13,14,15);
            const __bf16* p1 = p0 + 16 * 32;
            lo = *reinterpret_cast<const v8bf*>(p1);
            hi = *reinterpret_cast<const v8bf*>(p1 + 16);
            b1 = __builtin_shufflevector(lo, hi, 0,1,2,3,4,5,6,7,8,9,10,11,12,13,14,15);
        }

        acc00 = __builtin_amdgcn_wmma_f32_16x16x32_bf16(false, a0, false, b0, (short)0, acc00, false, false);
        acc01 = __builtin_amdgcn_wmma_f32_16x16x32_bf16(false, a0, false, b1, (short)0, acc01, false, false);
        acc10 = __builtin_amdgcn_wmma_f32_16x16x32_bf16(false, a1, false, b0, (short)0, acc10, false, false);
        acc11 = __builtin_amdgcn_wmma_f32_16x16x32_bf16(false, a1, false, b1, (short)0, acc11, false, false);

        if (hasNext) storeTile(1 - cur);       // write next buffer (read of it was last iter, before the barrier)
        __syncthreads();
    }

    // ---- epilogue ----
    const int mBase = m0 + waveM * 32 + half * 8;
    const int nBase = c0 + waveN * 32 + l15;
#pragma unroll
    for (int i = 0; i < 2; ++i) {
#pragma unroll
        for (int j = 0; j < 2; ++j) {
            const v8f accv = (i == 0) ? (j == 0 ? acc00 : acc01)
                                      : (j == 0 ? acc10 : acc11);
            const int n = nBase + j * 16;
            const float bi = bias ? bias[n] : 0.f;
#pragma unroll
            for (int r = 0; r < 8; ++r) {
                const int m = mBase + i * 16 + r;
                float vv = accv[r] + bi;
                if (doRelu)   vv = fmaxf(vv, 0.f);
                if (rowScale) vv *= rowScale[(long)m * rsStride];
                const long o = (long)m * ldc + n;
                if (accum) C[o] += vv; else C[o] = vv;
            }
        }
    }
}

// ---------------- embedding gather ----------------
__global__ __launch_bounds__(256)
void embed_kernel(const float* __restrict__ emb, const int* __restrict__ idx,
                  float* __restrict__ out, long n)
{
    long i = (long)blockIdx.x * 256 + threadIdx.x;
    if (i >= n) return;
    long t = i / DD;
    int  d = (int)(i % DD);
    out[i] = emb[(long)idx[t] * DD + d];
}

// ---------------- RoPE (tables recomputed on device) ----------------
__global__ __launch_bounds__(256)
void rope_kernel(float* __restrict__ x, long total)
{
    long i = (long)blockIdx.x * 256 + threadIdx.x;
    if (i >= total) return;
    const int  j   = (int)(i & 31);
    const long grp = i >> 5;                       // (b*S + s)*H + h
    const int  s   = (int)((grp / HH) % SS);
    float* base = x + grp * (long)HDIM;
    const float inv = __expf(-(float)j * (logf(10000.f) / 32.f));  // 10000^{-2j/64}
    const float ang = (float)s * inv;
    const float c = cosf(ang), sn = sinf(ang);
    const float x1 = base[j], x2 = base[j + 32];
    base[j]      = x1 * c - x2 * sn;
    base[j + 32] = x2 * c + x1 * sn;
}

// ---------------- scale + causal mask + softmax (one block per row) ----------------
__global__ __launch_bounds__(256)
void softmax_kernel(float* __restrict__ scores, int T, int Srows, float scale, int causal)
{
    const long row = blockIdx.x;
    const int  m   = (int)(row % Srows);
    float* p = scores + row * (long)T;
    __shared__ float red[256];
    const int tid = threadIdx.x;

    float lmax = -3.0e38f;
    for (int t = tid; t < T; t += 256) {
        float vv = p[t] * scale;
        if (causal && t > m) vv = -1.0e9f;
        p[t] = vv;
        lmax = fmaxf(lmax, vv);
    }
    red[tid] = lmax; __syncthreads();
    for (int s2 = 128; s2 > 0; s2 >>= 1) {
        if (tid < s2) red[tid] = fmaxf(red[tid], red[tid + s2]);
        __syncthreads();
    }
    const float mx = red[0]; __syncthreads();

    float lsum = 0.f;
    for (int t = tid; t < T; t += 256) { float e = __expf(p[t] - mx); p[t] = e; lsum += e; }
    red[tid] = lsum; __syncthreads();
    for (int s2 = 128; s2 > 0; s2 >>= 1) {
        if (tid < s2) red[tid] += red[tid + s2];
        __syncthreads();
    }
    const float inv = 1.f / red[0];
    for (int t = tid; t < T; t += 256) p[t] *= inv;
}

// ---------------- fused residual add + LayerNorm (one block per row of D=1024) ----------------
__global__ __launch_bounds__(256)
void add_ln_kernel(const float* __restrict__ x, const float* __restrict__ a,
                   const float* __restrict__ g, const float* __restrict__ bta,
                   float* __restrict__ out)
{
    __shared__ float vals[DD];
    __shared__ float red[256];
    const long row = blockIdx.x;
    const float* xr = x + row * (long)DD;
    const float* ar = a + row * (long)DD;
    const int tid = threadIdx.x;

    float s = 0.f;
    for (int d = tid; d < DD; d += 256) { float v = xr[d] + ar[d]; vals[d] = v; s += v; }
    red[tid] = s; __syncthreads();
    for (int st = 128; st > 0; st >>= 1) { if (tid < st) red[tid] += red[tid + st]; __syncthreads(); }
    const float mu = red[0] * (1.f / DD); __syncthreads();

    float vs = 0.f;
    for (int d = tid; d < DD; d += 256) { float t = vals[d] - mu; vs += t * t; }
    red[tid] = vs; __syncthreads();
    for (int st = 128; st > 0; st >>= 1) { if (tid < st) red[tid] += red[tid + st]; __syncthreads(); }
    const float inv = rsqrtf(red[0] * (1.f / DD) + 1e-5f);

    for (int d = tid; d < DD; d += 256)
        out[row * (long)DD + d] = (vals[d] - mu) * inv * g[d] + bta[d];
}

// ---------------- MoE top-2 gate: combine[tok, e] ----------------
__global__ __launch_bounds__(256)
void moe_gate_kernel(const float* __restrict__ x, const float* __restrict__ gw,
                     const float* __restrict__ gb, float* __restrict__ combine)
{
    const int wv   = threadIdx.x >> 5;
    const int lane = threadIdx.x & 31;
    const int tok  = blockIdx.x * 8 + wv;
    __shared__ float lg[8][EE];

    if (tok < BSTOK) {
        const int e = lane >> 2;      // expert 0..7
        const int part = lane & 3;    // 4-way split of the dot product
        const float* xr = x + (long)tok * DD;
        float s = 0.f;
        for (int d = part; d < DD; d += 4) s += xr[d] * gw[(long)d * EE + e];
        s += __shfl_xor(s, 1, 32);
        s += __shfl_xor(s, 2, 32);
        if (part == 0) lg[wv][e] = s + gb[e];
    }
    __syncthreads();

    if (tok < BSTOK && lane == 0) {
        float l[EE], p[EE], mx = -3.0e38f;
        for (int i = 0; i < EE; ++i) { l[i] = lg[wv][i]; mx = fmaxf(mx, l[i]); }
        float sum = 0.f;
        for (int i = 0; i < EE; ++i) { p[i] = __expf(l[i] - mx); sum += p[i]; }
        for (int i = 0; i < EE; ++i) p[i] /= sum;
        int i0 = 0;
        for (int i = 1; i < EE; ++i) if (p[i] > p[i0]) i0 = i;
        int i1 = (i0 == 0) ? 1 : 0;
        for (int i = 0; i < EE; ++i) if (i != i0 && p[i] > p[i1]) i1 = i;
        const float denom = p[i0] + p[i1];
        for (int i = 0; i < EE; ++i)
            combine[(long)tok * EE + i] =
                (i == i0) ? (p[i0] / denom) : ((i == i1) ? (p[i1] / denom) : 0.f);
    }
}

// ================= host-side orchestration =================
static inline void cvtL(hipStream_t st, const float* src, __bf16* dst, long n)
{
    const long n4 = n >> 2;   // all operand sizes here are multiples of 4
    cvt_bf16_kernel<<<(unsigned)((n4 + 255) / 256), 256, 0, st>>>(src, dst, n4);
}

static inline void gemmL(hipStream_t st, const __bf16* A, const __bf16* Bm, const float* bias,
                         const float* rowScale, int rsStride, float* C,
                         int M, int N, int K, int lda, int ldb, int ldc,
                         int bT, int relu, int accum,
                         int batch = 1, int batchH = 1,
                         long sA0 = 0, long sA1 = 0, long sB0 = 0, long sB1 = 0,
                         long sC0 = 0, long sC1 = 0)
{
    dim3 grid((M / 128) * (N / 64), batch);
    if (bT)
        gemm_wmma_kernel<1><<<grid, 256, 0, st>>>(A, Bm, bias, rowScale, rsStride, C,
                                                  M, N, K, lda, ldb, ldc, relu, accum,
                                                  batchH, sA0, sA1, sB0, sB1, sC0, sC1);
    else
        gemm_wmma_kernel<0><<<grid, 256, 0, st>>>(A, Bm, bias, rowScale, rsStride, C,
                                                  M, N, K, lda, ldb, ldc, relu, accum,
                                                  batchH, sA0, sA1, sB0, sB1, sC0, sC1);
}

// x/xkv: [B,S,D] f32; q/k/v/scores/attn/out: f32 scratch; abf/bbf: bf16 staging
static void run_mha(hipStream_t st, const float* xq, const float* xkv,
                    const float* wqkv, const float* bqkv,
                    const float* wo, const float* bo, int causal,
                    float* q, float* k, float* v, float* scores, float* attn, float* out,
                    __bf16* abf, __bf16* bbf)
{
    // q/k/v projections
    cvtL(st, xq, abf, (long)BSTOK * DD);
    cvtL(st, wqkv, bbf, (long)DD * DD);
    gemmL(st, abf, bbf, bqkv, nullptr, 0, q, BSTOK, DD, DD, DD, DD, DD, 0, 0, 0);
    if (xkv != xq) cvtL(st, xkv, abf, (long)BSTOK * DD);
    cvtL(st, wqkv + (long)DD * DD, bbf, (long)DD * DD);
    gemmL(st, abf, bbf, bqkv + DD, nullptr, 0, k, BSTOK, DD, DD, DD, DD, DD, 0, 0, 0);
    cvtL(st, wqkv + 2L * DD * DD, bbf, (long)DD * DD);
    gemmL(st, abf, bbf, bqkv + 2 * DD, nullptr, 0, v, BSTOK, DD, DD, DD, DD, DD, 0, 0, 0);

    const long nrot = (long)BSTOK * HH * 32;
    rope_kernel<<<(unsigned)((nrot + 255) / 256), 256, 0, st>>>(q, nrot);
    rope_kernel<<<(unsigned)((nrot + 255) / 256), 256, 0, st>>>(k, nrot);

    // scores[b,h] = q[b,h] (SxHD, lda=D) @ k[b,h]^T  (B transposed access)
    cvtL(st, q, abf, (long)BSTOK * DD);
    cvtL(st, k, bbf, (long)BSTOK * DD);
    gemmL(st, abf, bbf, nullptr, nullptr, 0, scores, SS, SS, HDIM, DD, DD, SS, 1, 0, 0,
          BB * HH, HH,
          (long)SS * DD, HDIM, (long)SS * DD, HDIM, (long)HH * SS * SS, (long)SS * SS);

    softmax_kernel<<<BB * HH * SS, 256, 0, st>>>(scores, SS, SS, 0.125f, causal);

    // attn[b,h] = softmax @ v[b,h]
    cvtL(st, scores, abf, (long)BB * HH * SS * SS);
    cvtL(st, v, bbf, (long)BSTOK * DD);
    gemmL(st, abf, bbf, nullptr, nullptr, 0, attn, SS, HDIM, SS, SS, DD, DD, 0, 0, 0,
          BB * HH, HH,
          (long)HH * SS * SS, (long)SS * SS, (long)SS * DD, HDIM, (long)SS * DD, HDIM);

    // out = attn @ wo + bo
    cvtL(st, attn, abf, (long)BSTOK * DD);
    cvtL(st, wo, bbf, (long)DD * DD);
    gemmL(st, abf, bbf, bo, nullptr, 0, out, BSTOK, DD, DD, DD, DD, DD, 0, 0, 0);
}

static void run_moe(hipStream_t st, const float* x,
                    const float* gw, const float* gb,
                    const float* w1, const float* b1,
                    const float* w2, const float* b2,
                    float* combine, float* hbuf, float* moeout,
                    __bf16* abf, __bf16* bbf, __bf16* hbf)
{
    moe_gate_kernel<<<BSTOK / 8, 256, 0, st>>>(x, gw, gb, combine);
    cvtL(st, x, abf, (long)BSTOK * DD);   // x in bf16, reused by all experts
    for (int e = 0; e < EE; ++e) {
        // h = relu(x @ w1[e] + b1[e])
        cvtL(st, w1 + (long)e * DD * FFD, bbf, (long)DD * FFD);
        gemmL(st, abf, bbf, b1 + (long)e * FFD, nullptr, 0,
              hbuf, BSTOK, FFD, DD, DD, FFD, FFD, 0, 1, 0);
        // moeout (+)= combine[:,e] * (h @ w2[e] + b2[e])
        cvtL(st, hbuf, hbf, (long)BSTOK * FFD);
        cvtL(st, w2 + (long)e * FFD * DD, bbf, (long)FFD * DD);
        gemmL(st, hbf, bbf, b2 + (long)e * DD, combine + e, EE,
              moeout, BSTOK, DD, FFD, FFD, DD, DD, 0, 0, (e > 0) ? 1 : 0);
    }
}

extern "C" void kernel_launch(void* const* d_in, const int* in_sizes, int n_in,
                              void* d_out, int out_size, void* d_ws, size_t ws_size,
                              hipStream_t stream)
{
    (void)in_sizes; (void)n_in; (void)out_size; (void)ws_size;
    int ii = 0;
    const int*   src        = (const int*)  d_in[ii++];
    const int*   tgt        = (const int*)  d_in[ii++];
    const float* emb_src    = (const float*)d_in[ii++];
    const float* emb_tgt    = (const float*)d_in[ii++];
    const float* enc_wqkv   = (const float*)d_in[ii++];
    const float* enc_bqkv   = (const float*)d_in[ii++];
    const float* enc_wo     = (const float*)d_in[ii++];
    const float* enc_bo     = (const float*)d_in[ii++];
    const float* enc_gate_w = (const float*)d_in[ii++];
    const float* enc_gate_b = (const float*)d_in[ii++];
    const float* enc_w1     = (const float*)d_in[ii++];
    const float* enc_b1     = (const float*)d_in[ii++];
    const float* enc_w2     = (const float*)d_in[ii++];
    const float* enc_b2     = (const float*)d_in[ii++];
    const float* enc_ln     = (const float*)d_in[ii++];
    const float* dec_self_wqkv  = (const float*)d_in[ii++];
    const float* dec_self_bqkv  = (const float*)d_in[ii++];
    const float* dec_self_wo    = (const float*)d_in[ii++];
    const float* dec_self_bo    = (const float*)d_in[ii++];
    const float* dec_cross_wqkv = (const float*)d_in[ii++];
    const float* dec_cross_bqkv = (const float*)d_in[ii++];
    const float* dec_cross_wo   = (const float*)d_in[ii++];
    const float* dec_cross_bo   = (const float*)d_in[ii++];
    const float* dec_gate_w = (const float*)d_in[ii++];
    const float* dec_gate_b = (const float*)d_in[ii++];
    const float* dec_w1     = (const float*)d_in[ii++];
    const float* dec_b1     = (const float*)d_in[ii++];
    const float* dec_w2     = (const float*)d_in[ii++];
    const float* dec_b2     = (const float*)d_in[ii++];
    const float* dec_ln     = (const float*)d_in[ii++];
    const float* final_w    = (const float*)d_in[ii++];
    const float* final_b    = (const float*)d_in[ii++];

    // scratch layout (float units; every size is a multiple of 4 -> 16B alignment holds)
    float* ws = (float*)d_ws;
    long off = 0;
    auto take = [&](long n) { float* p = ws + off; off += n; return p; };
    float* x       = take((long)BSTOK * DD);
    float* y       = take((long)BSTOK * DD);
    float* q       = take((long)BSTOK * DD);
    float* k       = take((long)BSTOK * DD);
    float* v       = take((long)BSTOK * DD);
    float* scores  = take((long)BB * HH * SS * SS);
    float* attn    = take((long)BSTOK * DD);
    float* proj    = take((long)BSTOK * DD);
    float* hbuf    = take((long)BSTOK * FFD);
    float* moeout  = take((long)BSTOK * DD);
    float* combine = take((long)BSTOK * EE);
    __bf16* abf = (__bf16*)take(((long)BB * HH * SS * SS) / 2);  // biggest A: scores
    __bf16* bbf = (__bf16*)take(((long)DD * VT) / 2);            // biggest B: final_w
    __bf16* hbf = (__bf16*)take(((long)BSTOK * FFD) / 2);        // MoE hidden

    const long nemb = (long)BSTOK * DD;
    embed_kernel<<<(unsigned)((nemb + 255) / 256), 256, 0, stream>>>(emb_src, src, x, nemb);
    embed_kernel<<<(unsigned)((nemb + 255) / 256), 256, 0, stream>>>(emb_tgt, tgt, y, nemb);

    // ---- encoder ----
    for (int l = 0; l < LL; ++l) {
        run_mha(stream, x, x,
                enc_wqkv + (long)l * 3 * DD * DD, enc_bqkv + (long)l * 3 * DD,
                enc_wo + (long)l * DD * DD, enc_bo + (long)l * DD, 0,
                q, k, v, scores, attn, proj, abf, bbf);
        add_ln_kernel<<<BSTOK, 256, 0, stream>>>(x, proj,
            enc_ln + ((long)(l * 2 + 0) * 2 + 0) * DD,
            enc_ln + ((long)(l * 2 + 0) * 2 + 1) * DD, x);

        run_moe(stream, x,
                enc_gate_w + (long)l * DD * EE, enc_gate_b + (long)l * EE,
                enc_w1 + (long)l * EE * DD * FFD, enc_b1 + (long)l * EE * FFD,
                enc_w2 + (long)l * EE * FFD * DD, enc_b2 + (long)l * EE * DD,
                combine, hbuf, moeout, abf, bbf, hbf);
        add_ln_kernel<<<BSTOK, 256, 0, stream>>>(x, moeout,
            enc_ln + ((long)(l * 2 + 1) * 2 + 0) * DD,
            enc_ln + ((long)(l * 2 + 1) * 2 + 1) * DD, x);
    }

    // ---- decoder ----
    for (int l = 0; l < LL; ++l) {
        run_mha(stream, y, y,
                dec_self_wqkv + (long)l * 3 * DD * DD, dec_self_bqkv + (long)l * 3 * DD,
                dec_self_wo + (long)l * DD * DD, dec_self_bo + (long)l * DD, 1,
                q, k, v, scores, attn, proj, abf, bbf);
        add_ln_kernel<<<BSTOK, 256, 0, stream>>>(y, proj,
            dec_ln + ((long)(l * 3 + 0) * 2 + 0) * DD,
            dec_ln + ((long)(l * 3 + 0) * 2 + 1) * DD, y);

        run_mha(stream, y, x,
                dec_cross_wqkv + (long)l * 3 * DD * DD, dec_cross_bqkv + (long)l * 3 * DD,
                dec_cross_wo + (long)l * DD * DD, dec_cross_bo + (long)l * DD, 0,
                q, k, v, scores, attn, proj, abf, bbf);
        add_ln_kernel<<<BSTOK, 256, 0, stream>>>(y, proj,
            dec_ln + ((long)(l * 3 + 1) * 2 + 0) * DD,
            dec_ln + ((long)(l * 3 + 1) * 2 + 1) * DD, y);

        run_moe(stream, y,
                dec_gate_w + (long)l * DD * EE, dec_gate_b + (long)l * EE,
                dec_w1 + (long)l * EE * DD * FFD, dec_b1 + (long)l * EE * FFD,
                dec_w2 + (long)l * EE * FFD * DD, dec_b2 + (long)l * EE * DD,
                combine, hbuf, moeout, abf, bbf, hbf);
        add_ln_kernel<<<BSTOK, 256, 0, stream>>>(y, moeout,
            dec_ln + ((long)(l * 3 + 2) * 2 + 0) * DD,
            dec_ln + ((long)(l * 3 + 2) * 2 + 1) * DD, y);
    }

    // ---- final vocab projection: [2048,1024] x [1024,32000] ----
    cvtL(stream, y, abf, (long)BSTOK * DD);
    cvtL(stream, final_w, bbf, (long)DD * VT);
    gemmL(stream, abf, bbf, final_b, nullptr, 0, (float*)d_out,
          BSTOK, VT, DD, DD, VT, VT, 0, 0, 0);
}
